// QuantizeLinear_83588653515161
// MI455X (gfx1250) — 8-bit matrix operand order repaired and checked by reading, compiled, not run on hardware
//
#include <hip/hip_runtime.h>
#include <cstdint>
#include <cstddef>

typedef __attribute__((ext_vector_type(8))) int          v8i;
typedef __attribute__((ext_vector_type(8))) float        v8f;
typedef __attribute__((ext_vector_type(4))) unsigned int u32x4;
typedef __attribute__((ext_vector_type(4))) int          i32x4;
typedef __attribute__((ext_vector_type(8))) int          i32x8;

#define IN_F   4096
#define OUT_F  4096
#define NGRP   32
#define GSIZE  128
#define MTOT   8192

#if defined(__has_builtin)
# if __has_builtin(__builtin_amdgcn_tensor_load_to_lds) && \
     __has_builtin(__builtin_amdgcn_s_wait_tensorcnt)
#  if defined(__has_include)
#   if __has_include(<hip/amd_detail/amd_gfx1250_TDM.h>)
#    define TDM_ARITY 6
#   else
#    define TDM_ARITY 5
#   endif
#  else
#   define TDM_ARITY 5
#  endif
# endif
#endif

union V8U {
    v8i   v;
    uint2 d2[4];
    int4  d4[2];
};

__global__ __launch_bounds__(256)
void act_quant_kernel(const float* __restrict__ x,
                      int8_t* __restrict__ xq8,
                      float* __restrict__ rinv) {
    const int row = blockIdx.x;
    const int t   = threadIdx.x;
    const float* xr = x + (size_t)row * IN_F;

    __shared__ float red[256];

    float4 v[4];
    float m = 0.0f;
#pragma unroll
    for (int j = 0; j < 4; ++j) {
        v[j] = ((const float4*)xr)[t * 4 + j];
        m = fmaxf(m, fmaxf(fmaxf(fabsf(v[j].x), fabsf(v[j].y)),
                           fmaxf(fabsf(v[j].z), fabsf(v[j].w))));
    }
    red[t] = m;
    __syncthreads();
#pragma unroll
    for (int s = 128; s > 0; s >>= 1) {
        if (t < s) red[t] = fmaxf(red[t], red[t + s]);
        __syncthreads();
    }
    const float amax = red[0];
    const float s    = 127.0f / (amax + 1e-6f);
    if (t == 0) rinv[row] = 1.0f / (s + 1e-6f);

    uint32_t pk[4];
#pragma unroll
    for (int j = 0; j < 4; ++j) {
        const int a = __float2int_rn(v[j].x * s) & 255;
        const int b = __float2int_rn(v[j].y * s) & 255;
        const int c = __float2int_rn(v[j].z * s) & 255;
        const int d = __float2int_rn(v[j].w * s) & 255;
        pk[j] = (uint32_t)(a | (b << 8) | (c << 16) | (d << 24));
    }
    uint4 o; o.x = pk[0]; o.y = pk[1]; o.z = pk[2]; o.w = pk[3];
    ((uint4*)(xq8 + (size_t)row * IN_F))[t] = o;
}

__global__ __launch_bounds__(256)
void wpack_kernel(const int* __restrict__ qweight,
                  const int* __restrict__ qzeros,
                  int8_t* __restrict__ w8t) {
    __shared__ int8_t lds[64 * 80];

    const int t  = threadIdx.x;
    const int k0 = blockIdx.x * 64;
    const int n0 = blockIdx.y * 64;

    const int nl = t & 63;
    const int kb = (t >> 6) * 16;
    const int z  = qzeros[(size_t)(k0 >> 7) * OUT_F + n0 + nl];

#pragma unroll
    for (int i = 0; i < 16; ++i) {
        const int q = qweight[(size_t)(k0 + kb + i) * OUT_F + n0 + nl];
        lds[nl * 80 + kb + i] = (int8_t)(q - z);
    }
    __syncthreads();

    const int nr = t >> 2;
    const int ks = (t & 3) * 16;
    const int4 val = *(const int4*)(lds + nr * 80 + ks);
    *(int4*)(w8t + (size_t)(n0 + nr) * IN_F + k0 + ks) = val;
}

#ifdef TDM_ARITY
__device__ __forceinline__ void tdm_load_2d(const void* gptr,
                                            unsigned lds_off,
                                            int tile_rows) {
    const unsigned long long ga = (unsigned long long)(size_t)gptr;
    u32x4 g0;
    g0.x = 1u;
    g0.y = lds_off;
    g0.z = (unsigned)ga;
    g0.w = ((unsigned)(ga >> 32) & 0x01FFFFFFu)
         | (2u << 30);
    i32x8 g1;
    g1[0] = (1 << 20)
          | (4 << 22)
          | (3 << 25);
    g1[1] = (int)(4096u << 16);
    g1[2] = 0;
    g1[3] = (int)((128u << 16) | 0x10u);
    g1[4] = tile_rows & 0xFFFF;
    g1[5] = 4096;
    g1[6] = 0;
    g1[7] = 0;
    const i32x4 z4 = {0, 0, 0, 0};
#if TDM_ARITY == 6
    const i32x8 z8 = {0, 0, 0, 0, 0, 0, 0, 0};
    __builtin_amdgcn_tensor_load_to_lds(g0, g1, z4, z4, z8, 0);
#else
    __builtin_amdgcn_tensor_load_to_lds(g0, g1, z4, z4, 0);
#endif
}
#endif

#define LDA 144
#define LDB 144

__global__ __launch_bounds__(256)
void gemm_iu8_kernel(const int8_t* __restrict__ xq8,
                     const int8_t* __restrict__ w8t,
                     const float* __restrict__ wscal,
                     const float* __restrict__ rinv,
                     float* __restrict__ out) {
    __shared__ int8_t lA[2][64  * LDA];
    __shared__ int8_t lB[2][128 * LDB];

    const int t    = threadIdx.x;
    const int lane = t & 31;
    const int wave = t >> 5;
    const int half = lane >> 4;
    const int r    = lane & 15;
    const int wm   = wave >> 2;
    const int wn   = wave & 3;

    const int m0 = blockIdx.y * 64;
    const int n0 = blockIdx.x * 128;

    v8f accf[2][2] = {{{0,0,0,0,0,0,0,0}, {0,0,0,0,0,0,0,0}},
                      {{0,0,0,0,0,0,0,0}, {0,0,0,0,0,0,0,0}}};

#ifdef TDM_ARITY
    if (t < 32) {
        tdm_load_2d(xq8 + (size_t)m0 * IN_F, (unsigned)(size_t)&lA[0][0], 64);
        tdm_load_2d(w8t + (size_t)n0 * IN_F, (unsigned)(size_t)&lB[0][0], 128);
    }
#endif

    for (int g = 0; g < NGRP; ++g) {
        const int cur = g & 1;
        const int kb  = g * GSIZE;

#ifdef TDM_ARITY
        if (g + 1 < NGRP) {
            if (t < 32) {
                tdm_load_2d(xq8 + (size_t)m0 * IN_F + kb + GSIZE,
                            (unsigned)(size_t)&lA[cur ^ 1][0], 64);
                tdm_load_2d(w8t + (size_t)n0 * IN_F + kb + GSIZE,
                            (unsigned)(size_t)&lB[cur ^ 1][0], 128);
            }
            __builtin_amdgcn_s_wait_tensorcnt(2);
        } else {
            __builtin_amdgcn_s_wait_tensorcnt(0);
        }
#else
#pragma unroll
        for (int p = 0; p < 2; ++p) {
            const int idx = p * 256 + t;
            const int row = idx >> 3;
            const int c   = (idx & 7) * 16;
            *(int4*)(&lA[cur][0] + row * LDA + c) =
                *(const int4*)(xq8 + (size_t)(m0 + row) * IN_F + kb + c);
        }
#pragma unroll
        for (int p = 0; p < 4; ++p) {
            const int idx = p * 256 + t;
            const int row = idx >> 3;
            const int c   = (idx & 7) * 16;
            *(int4*)(&lB[cur][0] + row * LDB + c) =
                *(const int4*)(w8t + (size_t)(n0 + row) * IN_F + kb + c);
        }
        if (g + 1 < NGRP) {
            __builtin_prefetch(xq8 + (size_t)(m0 + (t >> 2)) * IN_F + kb + GSIZE, 0, 1);
            __builtin_prefetch(w8t + (size_t)(n0 + (t >> 1)) * IN_F + kb + GSIZE, 0, 1);
        }
#endif
        __syncthreads();

        v8i acci[2][2] = {{{0,0,0,0,0,0,0,0}, {0,0,0,0,0,0,0,0}},
                          {{0,0,0,0,0,0,0,0}, {0,0,0,0,0,0,0,0}}};

#pragma unroll
        for (int kc = 0; kc < GSIZE; kc += 64) {
            V8U af[2];
#pragma unroll
            for (int mt = 0; mt < 2; ++mt) {
                const int8_t* base =
                    &lA[cur][0] + (wm * 32 + mt * 16 + r) * LDA + kc + half * 16;
                af[mt].d2[0] = *(const uint2*)(base + 0);
                af[mt].d2[1] = *(const uint2*)(base + 8);
                af[mt].d2[2] = *(const uint2*)(base + 32);
                af[mt].d2[3] = *(const uint2*)(base + 40);
            }
            V8U bf[2];
#pragma unroll
            for (int nt = 0; nt < 2; ++nt) {
                const int8_t* base =
                    &lB[cur][0] + (wn * 32 + nt * 16 + r) * LDB + kc + half * 16;
                bf[nt].d4[0] = *(const int4*)(base + 0);
                bf[nt].d4[1] = *(const int4*)(base + 32);
            }
#pragma unroll
            for (int mt = 0; mt < 2; ++mt)
#pragma unroll
                for (int nt = 0; nt < 2; ++nt)
                    acci[mt][nt] = __builtin_amdgcn_wmma_i32_16x16x64_iu8(
                         true, af[mt].v,
                         true, bf[nt].v,
                        acci[mt][nt],
                         false,  false);
        }

        float sc[2];
        sc[0] = wscal[(size_t)g * OUT_F + n0 + wn * 32 + r];
        sc[1] = wscal[(size_t)g * OUT_F + n0 + wn * 32 + 16 + r];
#pragma unroll
        for (int mt = 0; mt < 2; ++mt)
#pragma unroll
            for (int nt = 0; nt < 2; ++nt)
#pragma unroll
                for (int j = 0; j < 8; ++j)
                    accf[mt][nt][j] += sc[nt] * (float)acci[mt][nt][j];

        __syncthreads();
    }

#pragma unroll
    for (int mt = 0; mt < 2; ++mt) {
#pragma unroll
        for (int j = 0; j < 8; ++j) {
            const int m = m0 + wm * 32 + mt * 16 + j + half * 8;
            const float rv = rinv[m];
#pragma unroll
            for (int nt = 0; nt < 2; ++nt) {
                out[(size_t)m * OUT_F + n0 + wn * 32 + nt * 16 + r] =
                    accf[mt][nt][j] * rv;
            }
        }
    }
}

extern "C" void kernel_launch(void* const* d_in, const int* in_sizes, int n_in,
                              void* d_out, int out_size, void* d_ws, size_t ws_size,
                              hipStream_t stream) {
    (void)in_sizes; (void)n_in; (void)out_size; (void)ws_size;

    const float* x       = (const float*)d_in[0];
    const int*   qweight = (const int*)d_in[1];
    const int*   qzeros  = (const int*)d_in[2];
    const float* wscal   = (const float*)d_in[4];

    float*  out = (float*)d_out;
    int8_t* xq8 = (int8_t*)d_ws;
    int8_t* w8t = xq8 + (size_t)MTOT * IN_F;
    float*  rv  = (float*)(w8t + (size_t)OUT_F * IN_F);

    act_quant_kernel<<<MTOT, 256, 0, stream>>>(x, xq8, rv);
    wpack_kernel<<<dim3(IN_F / 64, OUT_F / 64), 256, 0, stream>>>(qweight, qzeros, w8t);
    gemm_iu8_kernel<<<dim3(OUT_F / 128, MTOT / 64), 256, 0, stream>>>(xq8, w8t, wscal, rv, out);
}
